// HAGNET_66529043415294
// MI455X (gfx1250) — compile-verified, matrix operand order checked by reading, not run on hardware
//
#include <hip/hip_runtime.h>

// ---------------- problem constants ----------------
#define NG   2000           // n_genes
#define T1C  7              // n_timestamps - 1
#define EC   128000         // edges per graph
#define ESL  (EC + NG)      // edges + self loops

typedef __attribute__((ext_vector_type(16))) __bf16 v16bf;
typedef __attribute__((ext_vector_type(8)))  float  v8f;

__device__ __forceinline__ float sigm(float x) { return 1.f / (1.f + __expf(-x)); }
__device__ __forceinline__ float actf(float v, int act) {
    if (act == 1) return fmaxf(v, 0.f);
    if (act == 2) return sigm(v);
    return v;
}
// order-preserving float<->uint key for atomicMax-based segment max
__device__ __forceinline__ unsigned fkey(float f) {
    unsigned u = __float_as_uint(f);
    return (u & 0x80000000u) ? ~u : (u | 0x80000000u);
}
__device__ __forceinline__ float funkey(unsigned u) {
    return (u & 0x80000000u) ? __uint_as_float(u & 0x7fffffffu) : __uint_as_float(~u);
}
// one 16B-per-lane async DMA transfer global -> LDS (CDNA5, ASYNCcnt-tracked)
__device__ __forceinline__ unsigned ldsoff(const void* p) {
    return (unsigned)(size_t)p;           // low 32 bits of flat LDS addr = offset
}
__device__ __forceinline__ void async_b128(unsigned lds_off, const float* gp) {
    asm volatile("global_load_async_to_lds_b128 %0, %1, off"
                 :: "v"(lds_off), "v"(gp) : "memory");
}
__device__ __forceinline__ void wait_async0() {
    asm volatile("s_wait_asynccnt 0" ::: "memory");
}

// ---------------------------------------------------------------------------
// bf16 WMMA GEMM:  C[M,N] = act( A[M,K] @ op(B) + bias[N] )
//   BT=1: B is [N,K] row-major (x @ W.T style);  BT=0: B is [K,N] row-major.
// Requires M%16==0, N%16==0, K%16==0 (all big layers here qualify).
// 4 waves/block; each wave owns one 16x16 C tile (v_wmma_f32_16x16x32_bf16,
// f32 accumulate). ALL staging is global_load_async_to_lds_b128; the K%32==16
// remainder tile zero-fills the upper LDS half with float4 stores. The wave id
// is readfirstlane'd so every tile guard is a scalar branch (EXEC stays ~0,
// as WMMA requires).
// ---------------------------------------------------------------------------
template <int BT>
__global__ __launch_bounds__(128) void k_gemm_wmma(
    const float* __restrict__ A, const float* __restrict__ B,
    const float* __restrict__ bias, float* __restrict__ C,
    int M, int N, int K, int act)
{
    __shared__ __attribute__((aligned(16))) float As[4][16][32];
    __shared__ __attribute__((aligned(16))) float Bsh[32 * 16];
    const int tid  = threadIdx.x;
    const int lane = tid & 31;
    const int wave = __builtin_amdgcn_readfirstlane(tid >> 5);   // scalar
    const int half = lane >> 4;
    const int l    = lane & 15;
    const int tn = blockIdx.x * 16;
    const int tm = (blockIdx.y * 4 + wave) * 16;
    const bool rowok = (tm < M);          // scalar; tile fully valid if true

    v8f acc = {0.f, 0.f, 0.f, 0.f, 0.f, 0.f, 0.f, 0.f};

    for (int k0 = 0; k0 < K; k0 += 32) {
        const bool kfull = (k0 + 32 <= K);   // else exactly 16 remain (K%16==0)

        // ---- stage A tile (16x32 f32 per wave), pure async DMA ----
        if (rowok) {
            if (kfull) {
                if (k0 + 64 <= K)
                    __builtin_prefetch(&A[(size_t)tm * K + k0 + 32], 0, 1);
                #pragma unroll
                for (int it = 0; it < 4; ++it) {
                    int idx = it * 32 + lane;           // 128 float4 xfers
                    int r = idx >> 3, c4 = (idx & 7) * 4;
                    async_b128(ldsoff(&As[wave][r][c4]),
                               &A[(size_t)(tm + r) * K + k0 + c4]);
                }
            } else {
                #pragma unroll
                for (int it = 0; it < 2; ++it) {
                    int idx = it * 32 + lane;           // 0..63
                    int r = idx >> 2, c4 = (idx & 3) * 4;
                    *(float4*)&As[wave][r][16 + c4] = make_float4(0.f, 0.f, 0.f, 0.f);
                    async_b128(ldsoff(&As[wave][r][c4]),
                               &A[(size_t)(tm + r) * K + k0 + c4]);
                }
            }
        }
        // ---- stage B tile (32x16 f32, block cooperative, pure async DMA) ----
        // BT=1: K-major LDS layout Bsh[c*32+k]; BT=0: N-major Bsh[k*16+c].
        if (kfull) {
            if (BT) {
                int cc = tid >> 3, k4 = (tid & 7) * 4;
                async_b128(ldsoff(&Bsh[cc * 32 + k4]),
                           &B[(size_t)(tn + cc) * K + k0 + k4]);
            } else {
                int kk = tid >> 2, c4 = (tid & 3) * 4;
                async_b128(ldsoff(&Bsh[kk * 16 + c4]),
                           &B[(size_t)(k0 + kk) * N + tn + c4]);
            }
        } else if (wave < 2) {                // scalar guard, 64 threads
            int idx = wave * 32 + lane;       // 0..63
            if (BT) {
                int cc = idx >> 2, k4 = (idx & 3) * 4;
                *(float4*)&Bsh[cc * 32 + 16 + k4] = make_float4(0.f, 0.f, 0.f, 0.f);
                async_b128(ldsoff(&Bsh[cc * 32 + k4]),
                           &B[(size_t)(tn + cc) * K + k0 + k4]);
            } else {
                int kk = idx >> 2, c4 = (idx & 3) * 4;
                *(float4*)&Bsh[(16 + kk) * 16 + c4] = make_float4(0.f, 0.f, 0.f, 0.f);
                async_b128(ldsoff(&Bsh[kk * 16 + c4]),
                           &B[(size_t)(k0 + kk) * N + tn + c4]);
            }
        }
        wait_async0();          // this wave's async DMAs landed in LDS
        __syncthreads();        // all waves' tiles visible

        // fragments per ISA 16-bit layout (lane halves hold K +8 / +16 offsets)
        v16bf afrag, bfrag;
        #pragma unroll
        for (int v = 0; v < 8; ++v) {
            #pragma unroll
            for (int p = 0; p < 2; ++p) {
                int kk = (v < 4) ? (v * 2 + p + half * 8)
                                 : (16 + (v - 4) * 2 + p + half * 8);
                afrag[2 * v + p] = (__bf16)As[wave][l][kk];
                bfrag[2 * v + p] = (__bf16)(BT ? Bsh[l * 32 + kk] : Bsh[kk * 16 + l]);
            }
        }
        acc = __builtin_amdgcn_wmma_f32_16x16x32_bf16(
                  false, afrag, false, bfrag, (short)0, acc, false, false);
        __syncthreads();
    }
    // C layout: VGPR r -> row r (+8 for upper lane half), lane%16 -> col
    if (rowok) {
        float bv = bias ? bias[tn + l] : 0.f;
        #pragma unroll
        for (int r = 0; r < 8; ++r) {
            int row = tm + r + half * 8;
            C[(size_t)row * N + tn + l] = actf(acc[r] + bv, act);
        }
    }
}

// naive GEMM for the two tiny layers with K=8 / N=8 (B is [N,K], BT=1 style)
__global__ void k_gemm_small(const float* __restrict__ A, const float* __restrict__ B,
                             const float* __restrict__ bias, float* __restrict__ C,
                             int M, int N, int K, int act)
{
    int i = blockIdx.x * blockDim.x + threadIdx.x;
    if (i >= M * N) return;
    int m = i / N, nn = i % N;
    float s = bias ? bias[nn] : 0.f;
    for (int k = 0; k < K; ++k) s += A[(size_t)m * K + k] * B[(size_t)nn * K + k];
    C[i] = actf(s, act);
}

// ---------------- elementwise / utility ----------------
__global__ void k_zero(float* p, int n) {
    int i = blockIdx.x * blockDim.x + threadIdx.x;
    if (i < n) p[i] = 0.f;
}
__global__ void k_copy(const float* a, float* b, int n) {
    int i = blockIdx.x * blockDim.x + threadIdx.x;
    if (i < n) b[i] = a[i];
}
__global__ void k_bias_act(float* x, const float* __restrict__ b, int n, int H, int act) {
    int i = blockIdx.x * blockDim.x + threadIdx.x;
    if (i >= n * H) return;
    x[i] = actf(x[i] + b[i % H], act);
}

// ---------------- GCN ----------------
__global__ void k_gcn_deg(const int* __restrict__ dst, const float* __restrict__ w,
                          float* deg, int E, int n) {
    int k = blockIdx.x * blockDim.x + threadIdx.x;
    if (k >= E + n) return;
    int d    = (k < E) ? dst[k] : (k - E);
    float ww = (k < E) ? w[k] : 2.0f;      // improved self-loop weight
    atomicAdd(&deg[d], ww);
}
__global__ void k_dinv(float* deg, int n) {
    int i = blockIdx.x * blockDim.x + threadIdx.x;
    if (i < n) deg[i] = (deg[i] > 0.f) ? rsqrtf(deg[i]) : 0.f;
}
__global__ void k_gcn_scatter(const int* __restrict__ src, const int* __restrict__ dst,
                              const float* __restrict__ w, const float* __restrict__ dinv,
                              const float* __restrict__ xw, float* __restrict__ out,
                              int E, int n, int H) {
    int k = blockIdx.x, t = threadIdx.x;   // blockDim == H
    int s, d; float ww;
    if (k < E) { s = src[k]; d = dst[k]; ww = w[k]; }
    else       { s = k - E;  d = s;      ww = 2.0f; }
    float coef = dinv[s] * ww * dinv[d];
    atomicAdd(&out[(size_t)d * H + t], coef * xw[(size_t)s * H + t]);
}

// ---------------- GAT ----------------
__global__ void k_rowdot2(const float* __restrict__ h, const float* __restrict__ as,
                          const float* __restrict__ ad, float* es, float* ed, int n, int H) {
    int i = blockIdx.x * blockDim.x + threadIdx.x;
    if (i >= n) return;
    float s = 0.f, d = 0.f;
    for (int k = 0; k < H; ++k) { float v = h[(size_t)i * H + k]; s += v * as[k]; d += v * ad[k]; }
    es[i] = s; ed[i] = d;
}
__global__ void k_edge_max(const int* __restrict__ src, const int* __restrict__ dst,
                           const float* __restrict__ es, const float* __restrict__ ed,
                           unsigned* __restrict__ mkey, float* __restrict__ ev, int E, int n) {
    int k = blockIdx.x * blockDim.x + threadIdx.x;
    if (k >= E + n) return;
    int s = (k < E) ? src[k] : (k - E);
    int d = (k < E) ? dst[k] : (k - E);
    float e = es[s] + ed[d];
    e = (e > 0.f) ? e : 0.2f * e;          // leaky_relu(0.2)
    ev[k] = e;
    atomicMax(&mkey[d], fkey(e));
}
__global__ void k_edge_expsum(const int* __restrict__ dst, float* __restrict__ ev,
                              const unsigned* __restrict__ mkey, float* __restrict__ ssum,
                              int E, int n) {
    int k = blockIdx.x * blockDim.x + threadIdx.x;
    if (k >= E + n) return;
    int d = (k < E) ? dst[k] : (k - E);
    float ex = __expf(ev[k] - funkey(mkey[d]));
    ev[k] = ex;
    atomicAdd(&ssum[d], ex);
}
__global__ void k_edge_scatter(const int* __restrict__ src, const int* __restrict__ dst,
                               const float* __restrict__ ev, const float* __restrict__ ssum,
                               const float* __restrict__ h, float* __restrict__ out,
                               int E, int n, int H) {
    int k = blockIdx.x, t = threadIdx.x;   // blockDim == H
    int s = (k < E) ? src[k] : (k - E);
    int d = (k < E) ? dst[k] : (k - E);
    float alpha = ev[k] / ssum[d];
    atomicAdd(&out[(size_t)d * H + t], alpha * h[(size_t)s * H + t]);
}

// ---------------- TA: per-column BatchNorm over genes + sigmoid mask ----------------
__global__ __launch_bounds__(256) void k_ta_bn(
    const float* __restrict__ y, const float* __restrict__ cb,
    const float* __restrict__ g, const float* __restrict__ be,
    const float* __restrict__ xsj, float* __restrict__ out, int n, int H)
{
    int h = blockIdx.x;                    // one block per hidden channel
    __shared__ float s1[256], s2[256];
    float a = 0.f, b = 0.f;
    for (int r = threadIdx.x; r < n; r += 256) {
        float v = y[(size_t)r * H + h] + cb[r];
        a += v; b += v * v;
    }
    s1[threadIdx.x] = a; s2[threadIdx.x] = b; __syncthreads();
    for (int s = 128; s > 0; s >>= 1) {
        if (threadIdx.x < s) { s1[threadIdx.x] += s1[threadIdx.x + s];
                               s2[threadIdx.x] += s2[threadIdx.x + s]; }
        __syncthreads();
    }
    float mu  = s1[0] / (float)n;
    float var = s2[0] / (float)n - mu * mu;
    float inv = rsqrtf(var + 1e-5f);
    float gg = g[h], bb = be[h];
    for (int r = threadIdx.x; r < n; r += 256) {
        float v = y[(size_t)r * H + h] + cb[r];
        float mask = sigm((v - mu) * inv * gg + bb);
        out[(size_t)r * H + h] = xsj[(size_t)r * H + h] * mask;
    }
}

// ---------------- losses:  wm*MSE + wk*KL(batchmean) ----------------
__global__ __launch_bounds__(256) void k_loss(
    const float* __restrict__ inp, const float* __restrict__ tgt,
    float* accum, int total, int rows, float wm, float wk)
{
    __shared__ float red[256];
    int i = blockIdx.x * 256 + threadIdx.x;
    float v = 0.f;
    if (i < total) {
        float a = inp[i], t = tgt[i];
        v = wm * (a - t) * (a - t) / (float)total
          + wk * t * (__logf(t) - a) / (float)rows;
    }
    red[threadIdx.x] = v; __syncthreads();
    for (int s = 128; s > 0; s >>= 1) {
        if (threadIdx.x < s) red[threadIdx.x] += red[threadIdx.x + s];
        __syncthreads();
    }
    if (threadIdx.x == 0) atomicAdd(accum, red[0]);
}

// ---------------- output assembly ----------------
__global__ void k_concat(const float* __restrict__ xg, const float* __restrict__ xd,
                         float* __restrict__ out, int n) {
    int i = blockIdx.x * blockDim.x + threadIdx.x;
    if (i >= n * 96) return;
    int r = i / 96, c = i % 96;
    out[i] = (c < 32) ? xg[r * 32 + c] : xd[r * 64 + (c - 32)];
}
__global__ void k_finish(const float* acc, float* out) {
    if (threadIdx.x == 0) out[0] = acc[0];
}

// ===========================================================================
extern "C" void kernel_launch(void* const* d_in, const int* in_sizes, int n_in,
                              void* d_out, int out_size, void* d_ws, size_t ws_size,
                              hipStream_t stream)
{
    (void)in_sizes; (void)n_in; (void)out_size; (void)ws_size;
    const int n = NG, E = EC;

    // ---- inputs (setup_inputs order) ----
    const float* x_g     = (const float*)d_in[0];
    const int*   eg      = (const int*)  d_in[1];   // [2,E]
    const float* wg      = (const float*)d_in[2];
    const float* x_d     = (const float*)d_in[3];   // [7,n,256]
    const int*   ed_all  = (const int*)  d_in[4];   // [7,2,E]
    const float* gcn_W0  = (const float*)d_in[5];
    const float* gcn_b0  = (const float*)d_in[6];
    const float* gcn_W1  = (const float*)d_in[7];
    const float* gcn_b1  = (const float*)d_in[8];
    const float* dg_W0   = (const float*)d_in[9];
    const float* dg_b0   = (const float*)d_in[10];
    const float* dg_W1   = (const float*)d_in[11];
    const float* dg_b1   = (const float*)d_in[12];
    const float* gat_W0  = (const float*)d_in[13];  // [7,128,256]
    const float* gat_b0  = (const float*)d_in[14];
    const float* gat_as0 = (const float*)d_in[15];
    const float* gat_ad0 = (const float*)d_in[16];
    const float* gat_W1  = (const float*)d_in[17];  // [7,64,128]
    const float* gat_b1  = (const float*)d_in[18];
    const float* gat_as1 = (const float*)d_in[19];
    const float* gat_ad1 = (const float*)d_in[20];
    const float* ta_cw0  = (const float*)d_in[21];  // [6,n,n]
    const float* ta_cb0  = (const float*)d_in[22];
    const float* ta_g0   = (const float*)d_in[23];
    const float* ta_be0  = (const float*)d_in[24];
    const float* ta_cw1  = (const float*)d_in[25];  // [6,n,n]
    const float* ta_cb1  = (const float*)d_in[26];
    const float* ta_g1   = (const float*)d_in[27];
    const float* ta_be1  = (const float*)d_in[28];
    const float* dd_W0   = (const float*)d_in[29];  // [7,128,64]
    const float* dd_b0   = (const float*)d_in[30];
    const float* dd_W1   = (const float*)d_in[31];  // [7,256,128]
    const float* dd_b1   = (const float*)d_in[32];
    float* out = (float*)d_out;

    // ---- workspace bump allocator ----
    float* base = (float*)d_ws;
    size_t pos = 0;
    auto alloc = [&](size_t c) { float* p = base + pos; pos += c; return p; };
    float* sAcc   = alloc(16);
    float* deg    = alloc(n);
    float* xw     = alloc((size_t)n * 64);
    float* accA   = alloc((size_t)n * 64);
    float* xg_enc = alloc((size_t)n * 32);
    float* dgh    = alloc((size_t)n * 64);
    float* xg_rec = alloc((size_t)n * 8);
    float* xs0    = alloc((size_t)T1C * n * 128);
    float* xs1    = alloc((size_t)T1C * n * 64);
    float* taY    = alloc((size_t)n * 128);
    float* tempA  = alloc((size_t)n * 128);
    float* tempB  = alloc((size_t)n * 128);
    float* hbuf   = alloc((size_t)n * 128);
    float* es     = alloc(n);
    float* edv    = alloc(n);
    float* evale  = alloc(ESL);
    float* mkeyf  = alloc(n);
    float* ssum   = alloc(n);
    float* xd_enc = alloc((size_t)T1C * n * 64);
    float* ddh    = alloc((size_t)n * 128);
    float* xd_rec = alloc((size_t)T1C * n * 256);

    auto zero = [&](float* p, int c) {
        k_zero<<<(c + 255) / 256, 256, 0, stream>>>(p, c);
    };
    auto gemm = [&](const float* A, const float* Bm, const float* bias, float* C,
                    int M, int N, int K, int BT, int act) {
        if ((K % 16 == 0) && (N % 16 == 0)) {
            dim3 g(N / 16, (M + 63) / 64);
            if (BT) k_gemm_wmma<1><<<g, 128, 0, stream>>>(A, Bm, bias, C, M, N, K, act);
            else    k_gemm_wmma<0><<<g, 128, 0, stream>>>(A, Bm, bias, C, M, N, K, act);
        } else {  // only the two tiny BT=1 layers land here (K=8 or N=8)
            k_gemm_small<<<(M * N + 255) / 256, 256, 0, stream>>>(A, Bm, bias, C, M, N, K, act);
        }
    };
    auto gat_step = [&](const float* xin, int Hin, int Hout,
                        const float* W, const float* b,
                        const float* as, const float* ad,
                        const int* src, const int* dst, float* o, int act) {
        gemm(xin, W, nullptr, hbuf, n, Hout, Hin, 1, 0);
        k_rowdot2<<<(n + 255) / 256, 256, 0, stream>>>(hbuf, as, ad, es, edv, n, Hout);
        zero(mkeyf, n);
        k_edge_max<<<(ESL + 255) / 256, 256, 0, stream>>>(src, dst, es, edv,
                                                          (unsigned*)mkeyf, evale, E, n);
        zero(ssum, n);
        k_edge_expsum<<<(ESL + 255) / 256, 256, 0, stream>>>(dst, evale,
                                                             (unsigned*)mkeyf, ssum, E, n);
        zero(o, n * Hout);
        k_edge_scatter<<<ESL, Hout, 0, stream>>>(src, dst, evale, ssum, hbuf, o, E, n, Hout);
        k_bias_act<<<(n * Hout + 255) / 256, 256, 0, stream>>>(o, b, n, Hout, act);
    };

    zero(sAcc, 16);

    // ================= GCN encoder =================
    const int* eg_src = eg;
    const int* eg_dst = eg + E;
    zero(deg, n);
    k_gcn_deg<<<(ESL + 255) / 256, 256, 0, stream>>>(eg_dst, wg, deg, E, n);
    k_dinv<<<(n + 255) / 256, 256, 0, stream>>>(deg, n);
    // layer 1: h = relu(gcn(x_g, W0))
    gemm(x_g, gcn_W0, nullptr, xw, n, 64, 8, 1, 0);
    zero(accA, n * 64);
    k_gcn_scatter<<<ESL, 64, 0, stream>>>(eg_src, eg_dst, wg, deg, xw, accA, E, n, 64);
    k_bias_act<<<(n * 64 + 255) / 256, 256, 0, stream>>>(accA, gcn_b0, n, 64, 1);
    // layer 2: xg_enc = sigmoid(gcn(h, W1))
    gemm(accA, gcn_W1, nullptr, xw, n, 32, 64, 1, 0);
    zero(xg_enc, n * 32);
    k_gcn_scatter<<<ESL, 32, 0, stream>>>(eg_src, eg_dst, wg, deg, xw, xg_enc, E, n, 32);
    k_bias_act<<<(n * 32 + 255) / 256, 256, 0, stream>>>(xg_enc, gcn_b1, n, 32, 2);

    // ================= Decoder_G + loss_g =================
    gemm(xg_enc, dg_W0, dg_b0, dgh, n, 64, 32, 1, 1);
    gemm(dgh, dg_W1, dg_b1, xg_rec, n, 8, 64, 1, 2);
    k_loss<<<(n * 8 + 255) / 256, 256, 0, stream>>>(x_g, xg_rec, sAcc, n * 8, n, 1.f, 1.f);

    // ================= TA-GAT encoder =================
    // layer 0 (relu): inputs x_d[j], no TA mask
    for (int j = 0; j < T1C; ++j) {
        const int* src = ed_all + (size_t)j * 2 * E;
        const int* dst = src + E;
        gat_step(x_d + (size_t)j * n * 256, 256, 128,
                 gat_W0 + (size_t)j * 128 * 256, gat_b0 + (size_t)j * 128,
                 gat_as0 + (size_t)j * 128, gat_ad0 + (size_t)j * 128,
                 src, dst, xs0 + (size_t)j * n * 128, 1);
    }
    // layer 1 (sigmoid): TA mask from previous temp for j>=1
    {
        const int* src0 = ed_all;
        const int* dst0 = src0 + E;
        gat_step(xs0, 128, 64, gat_W1, gat_b1, gat_as1, gat_ad1,
                 src0, dst0, xs1, 2);
        const float* tprev = xs0;   // temp used at j=0
        float* ping = tempA; float* pong = tempB;
        for (int j = 1; j < T1C; ++j) {
            // y = ta_cw0[j-1] @ temp_prev   (WMMA-dominant matmul: 2000x2000x128)
            gemm(ta_cw0 + (size_t)(j - 1) * n * n, tprev, nullptr, taY, n, 128, n, 0, 0);
            k_ta_bn<<<128, 256, 0, stream>>>(taY, ta_cb0 + (size_t)(j - 1) * n,
                                             ta_g0 + (size_t)(j - 1) * 128,
                                             ta_be0 + (size_t)(j - 1) * 128,
                                             xs0 + (size_t)j * n * 128, ping, n, 128);
            const int* src = ed_all + (size_t)j * 2 * E;
            const int* dst = src + E;
            gat_step(ping, 128, 64,
                     gat_W1 + (size_t)j * 64 * 128, gat_b1 + (size_t)j * 64,
                     gat_as1 + (size_t)j * 64, gat_ad1 + (size_t)j * 64,
                     src, dst, xs1 + (size_t)j * n * 64, 2);
            tprev = ping; float* t = ping; ping = pong; pong = t;
        }
    }
    // final TA chain -> xd_enc
    k_copy<<<(n * 64 + 255) / 256, 256, 0, stream>>>(xs1, xd_enc, n * 64);
    {
        const float* tprev = xd_enc;
        for (int j = 0; j < T1C - 1; ++j) {
            gemm(ta_cw1 + (size_t)j * n * n, tprev, nullptr, taY, n, 64, n, 0, 0);
            k_ta_bn<<<64, 256, 0, stream>>>(taY, ta_cb1 + (size_t)j * n,
                                            ta_g1 + (size_t)j * 64,
                                            ta_be1 + (size_t)j * 64,
                                            xs1 + (size_t)(j + 1) * n * 64,
                                            xd_enc + (size_t)(j + 1) * n * 64, n, 64);
            tprev = xd_enc + (size_t)(j + 1) * n * 64;
        }
    }

    // ================= Decoder_D + loss_d + loss_t =================
    for (int t = 0; t < T1C; ++t) {
        gemm(xd_enc + (size_t)t * n * 64, dd_W0 + (size_t)t * 128 * 64,
             dd_b0 + (size_t)t * 128, ddh, n, 128, 64, 1, 1);
        gemm(ddh, dd_W1 + (size_t)t * 256 * 128,
             dd_b1 + (size_t)t * 256, xd_rec + (size_t)t * n * 256, n, 256, 128, 1, 2);
    }
    const int last = T1C - 1;
    k_loss<<<(n * 256 + 255) / 256, 256, 0, stream>>>(
        x_d + (size_t)last * n * 256, xd_rec + (size_t)last * n * 256,
        sAcc, n * 256, n, 1.f, 1.f);
    const float wkt = 0.5f / (float)T1C;   // LOSS_T_WEIGHT * (1/T1)
    for (int i = 0; i < T1C - 1; ++i) {
        k_loss<<<(n * 256 + 255) / 256, 256, 0, stream>>>(
            xd_rec + (size_t)i * n * 256, xd_rec + (size_t)(i + 1) * n * 256,
            sAcc, n * 256, n, 0.f, wkt);
    }

    // ================= outputs =================
    k_concat<<<(n * 96 + 255) / 256, 256, 0, stream>>>(
        xg_enc, xd_enc + (size_t)last * n * 64, out, n);
    k_finish<<<1, 32, 0, stream>>>(sAcc, out + (size_t)n * 96);
}
